// DeformableConv2_73538430042695
// MI455X (gfx1250) — matrix operand order repaired and checked by reading, not compiled, not run on hardware
//
#include <hip/hip_runtime.h>
#include <hip/hip_bf16.h>

typedef __attribute__((ext_vector_type(2)))  float  v2f;
typedef __attribute__((ext_vector_type(8)))  float  v8f;
typedef __attribute__((ext_vector_type(16))) __bf16 v16bf;
typedef __attribute__((ext_vector_type(8)))  __bf16 v8bf;

union BFrag { v16bf v; v8bf h[2]; };

#define CIN   768
#define COUT  768
#define HW    1024   // 32*32
#define NPIX  8192   // B*H*W
#define KDIM  6912   // CIN*9
#define CCH   32     // c-chunk (K-chunk = 288 = 9 steps of K=32)
#define KCH   (CCH * 9)

// ---------------------------------------------------------------------------
// Kernel 0: split proj_w (fp32) into bf16 hi/lo planes for bf16x3 WMMA.
// ---------------------------------------------------------------------------
__global__ __launch_bounds__(256) void k_wsplit(
    const float* __restrict__ wp, __bf16* __restrict__ whi, __bf16* __restrict__ wlo)
{
  size_t i = (size_t)blockIdx.x * 256 + threadIdx.x;
  if (i >= (size_t)COUT * KDIM) return;
  float v = wp[i];
  __bf16 h = (__bf16)v;
  whi[i] = h;
  wlo[i] = (__bf16)(v - (float)h);
}

// ---------------------------------------------------------------------------
// Kernel 1: offset conv (18ch) + mask conv (9ch) fused, then bilinear tables.
// One thread per (pixel n, tap k): 3 dot products of length 6912, then emits
// 4 corner (weight*mask, spatial index) pairs. fp32 exact.
// ---------------------------------------------------------------------------
__global__ __launch_bounds__(256) void k_offmask(
    const float* __restrict__ x,
    const float* __restrict__ ow, const float* __restrict__ ob,
    const float* __restrict__ mw, const float* __restrict__ mb,
    float* __restrict__ tw, int* __restrict__ tix)
{
  int t = blockIdx.x * 256 + threadIdx.x;
  if (t >= NPIX * 9) return;
  int n = t / 9, k = t - n * 9;
  int b = n >> 10, s = n & 1023, h = s >> 5, w = s & 31;
  int ki = k / 3, kj = k - ki * 3;

  const int oc_dy = 2 * k, oc_dx = 2 * k + 1;
  float ady = 0.f, adx = 0.f, am = 0.f;

  for (int ti = 0; ti < 3; ++ti) {
    int hh = h + ti - 1;
    if ((unsigned)hh >= 32u) continue;
    for (int tj = 0; tj < 3; ++tj) {
      int ww = w + tj - 1;
      if ((unsigned)ww >= 32u) continue;
      int sp = (hh << 5) + ww;
      const float* xp  = x  + (((size_t)b * CIN) << 10) + sp;
      const float* wdy = ow + (size_t)oc_dy * CIN * 9 + ti * 3 + tj;
      const float* wdx = ow + (size_t)oc_dx * CIN * 9 + ti * 3 + tj;
      const float* wm  = mw + (size_t)k     * CIN * 9 + ti * 3 + tj;
      for (int c = 0; c < CIN; ++c) {
        float xv = xp[(size_t)c << 10];
        ady = fmaf(xv, wdy[c * 9], ady);
        adx = fmaf(xv, wdx[c * 9], adx);
        am  = fmaf(xv, wm [c * 9], am);
      }
    }
  }

  float dy = fminf(fmaxf(ady + ob[oc_dy], -8.f), 8.f);   // max_off = 32/4 = 8
  float dx = fminf(fmaxf(adx + ob[oc_dx], -8.f), 8.f);
  float m  = 2.f / (1.f + expf(-(am + mb[k])));

  float ys = dy + (float)(h + ki - 1);
  float xs = dx + (float)(w + kj - 1);
  float y0f = floorf(ys), x0f = floorf(xs);
  float wy = ys - y0f, wx = xs - x0f;

  float cw[4] = { (1.f - wy) * (1.f - wx), (1.f - wy) * wx,
                  wy * (1.f - wx),         wy * wx };
  float cy[4] = { y0f, y0f, y0f + 1.f, y0f + 1.f };
  float cx[4] = { x0f, x0f + 1.f, x0f, x0f + 1.f };

  int base = t * 4;
  #pragma unroll
  for (int j = 0; j < 4; ++j) {
    bool valid = (cy[j] >= 0.f) && (cy[j] <= 31.f) &&
                 (cx[j] >= 0.f) && (cx[j] <= 31.f);
    int yc = (int)fminf(fmaxf(cy[j], 0.f), 31.f);
    int xc = (int)fminf(fmaxf(cx[j], 0.f), 31.f);
    tw [base + j] = valid ? cw[j] * m : 0.f;
    tix[base + j] = (yc << 5) + xc;
  }
}

// ---------------------------------------------------------------------------
// Kernel 2: deformable-conv GEMM via V_WMMA_F32_16X16X32_BF16, bf16x3 split:
//   D += Ahi*Bhi + Ahi*Blo + Alo*Bhi   (f32 accumulate, ~fp32 accuracy)
// Workgroup = 512 threads (16 wave32). Tile: full M=768 x N=32 pixels, so the
// gathered matrix S is built exactly once per element. K loop: c-chunks of 32
// (K-chunk 288 = 9 WMMA steps). Wave wv owns M-tiles {3wv..3wv+2} x N-tiles
// {0,1} -> 6 v8f accumulators.
// ---------------------------------------------------------------------------
__global__ __launch_bounds__(512) void k_dcnv_gemm(
    const float* __restrict__ x,
    const __bf16* __restrict__ whi, const __bf16* __restrict__ wlo,
    const float* __restrict__ pb,
    const float* __restrict__ tw, const int* __restrict__ tix,
    float* __restrict__ y)
{
  __shared__ __bf16 sHi[32 * KCH];   // S slab, hi plane: row-per-pixel [n][K]
  __shared__ __bf16 sLo[32 * KCH];   // S slab, lo plane
  __shared__ float  sTw[32 * 9 * 4];
  __shared__ int    sTi[32 * 9 * 4];

  const int tid   = threadIdx.x;
  const int lane  = tid & 31;
  const int wv    = tid >> 5;         // 0..15
  const int mrow  = lane & 15;        // M (A) / N (B,C) position
  const int half  = lane >> 4;        // half-wave select

  const int n0 = blockIdx.x * 32;     // 32 pixels, never crosses batch bound
  const int b  = n0 >> 10;
  const int s0 = n0 & 1023;

  for (int t = tid; t < 32 * 9 * 4; t += 512) {
    sTw[t] = tw [(size_t)n0 * 36 + t];
    sTi[t] = tix[(size_t)n0 * 36 + t];
  }

  v8f acc[6] = {};

  for (int cc = 0; cc < CIN; cc += CCH) {
    __syncthreads();
    // ---- gather S slab + bf16 hi/lo split; lanes write contiguous bf16 ----
    for (int t = tid; t < 32 * KCH; t += 512) {
      int nl = t / KCH, r = t - nl * KCH;     // r = cl*9 + k in [0,288)
      int cl = r / 9, k = r - cl * 9;
      const float* xb = x + (((size_t)(b * CIN + cc + cl)) << 10);
      const float* w4 = &sTw[(nl * 9 + k) * 4];
      const int*   i4 = &sTi[(nl * 9 + k) * 4];
      float v = w4[0] * xb[i4[0]] + w4[1] * xb[i4[1]] +
                w4[2] * xb[i4[2]] + w4[3] * xb[i4[3]];
      __bf16 hv = (__bf16)v;
      sHi[nl * KCH + r] = hv;
      sLo[nl * KCH + r] = (__bf16)(v - (float)hv);
    }
    __syncthreads();

    if (cc + CCH < CIN)  // warm next weight slab (global_prefetch_b8)
      __builtin_prefetch(whi + (size_t)(wv * 48 + mrow) * KDIM + (cc + CCH) * 9, 0, 1);

    for (int ks = 0; ks < 9; ++ks) {          // 9 * K32 = 288
      // B fragments (32x16 bf16): lane = col N=mrow(+16), K in
      // [ks*32+16*half, +16) -> 16 contiguous bf16 = 2x ds_load_b128
      const int kbB = ks * 32 + 16 * half;
      BFrag b0h, b0l, b1h, b1l;
      {
        const __bf16* p0 = &sHi[mrow * KCH + kbB];
        const __bf16* p1 = &sHi[(mrow + 16) * KCH + kbB];
        const __bf16* q0 = &sLo[mrow * KCH + kbB];
        const __bf16* q1 = &sLo[(mrow + 16) * KCH + kbB];
        b0h.h[0] = *(const v8bf*)p0;       b0h.h[1] = *(const v8bf*)(p0 + 8);
        b1h.h[0] = *(const v8bf*)p1;       b1h.h[1] = *(const v8bf*)(p1 + 8);
        b0l.h[0] = *(const v8bf*)q0;       b0l.h[1] = *(const v8bf*)(q0 + 8);
        b1l.h[0] = *(const v8bf*)q1;       b1l.h[1] = *(const v8bf*)(q1 + 8);
      }
      #pragma unroll
      for (int j2 = 0; j2 < 3; ++j2) {
        int mt = wv * 3 + j2;
        // A fragment (16x32 bf16): lane row M=mrow, K runs [16*half,+16)
        // contiguous -> 2x global_load_b128 per plane
        const size_t abase = (size_t)(mt * 16 + mrow) * KDIM
                           + (size_t)cc * 9 + ks * 32 + 16 * half;
        BFrag ah, al;
        ah.h[0] = *(const v8bf*)(whi + abase);
        ah.h[1] = *(const v8bf*)(whi + abase + 8);
        al.h[0] = *(const v8bf*)(wlo + abase);
        al.h[1] = *(const v8bf*)(wlo + abase + 8);

        v8f a0 = acc[j2 * 2 + 0];
        a0 = __builtin_amdgcn_wmma_f32_16x16x32_bf16(false, ah.v, false, b0h.v, (short)0, a0, false, false);
        a0 = __builtin_amdgcn_wmma_f32_16x16x32_bf16(false, ah.v, false, b0l.v, (short)0, a0, false, false);
        a0 = __builtin_amdgcn_wmma_f32_16x16x32_bf16(false, al.v, false, b0h.v, (short)0, a0, false, false);
        acc[j2 * 2 + 0] = a0;

        v8f a1 = acc[j2 * 2 + 1];
        a1 = __builtin_amdgcn_wmma_f32_16x16x32_bf16(false, ah.v, false, b1h.v, (short)0, a1, false, false);
        a1 = __builtin_amdgcn_wmma_f32_16x16x32_bf16(false, ah.v, false, b1l.v, (short)0, a1, false, false);
        a1 = __builtin_amdgcn_wmma_f32_16x16x32_bf16(false, al.v, false, b1h.v, (short)0, a1, false, false);
        acc[j2 * 2 + 1] = a1;
      }
    }
  }

  // ---- write D + bias; C/D layout: VGPR e, lane -> M = e + 8*half, N = mrow
  #pragma unroll
  for (int j2 = 0; j2 < 3; ++j2) {
    int mt = wv * 3 + j2;
    #pragma unroll
    for (int nt = 0; nt < 2; ++nt) {
      v8f a = acc[j2 * 2 + nt];
      #pragma unroll
      for (int e = 0; e < 8; ++e) {
        int o = mt * 16 + e + 8 * half;
        int s = s0 + nt * 16 + mrow;
        y[(((size_t)(b * COUT + o)) << 10) + s] = a[e] + pb[o];
      }
    }
  }
}

// ---------------------------------------------------------------------------
// Kernel 3: per-channel mean / rsqrt(var + eps) over (B,H,W).
// ---------------------------------------------------------------------------
__global__ __launch_bounds__(256) void k_stats(
    const float* __restrict__ y, float* __restrict__ stats)
{
  __shared__ float s1[256], s2[256];
  int o = blockIdx.x;
  float sum = 0.f, sum2 = 0.f;
  for (int i = threadIdx.x; i < NPIX; i += 256) {
    int bb = i >> 10, s = i & 1023;
    float v = y[(((size_t)(bb * COUT + o)) << 10) + s];
    sum += v; sum2 += v * v;
  }
  s1[threadIdx.x] = sum; s2[threadIdx.x] = sum2;
  __syncthreads();
  for (int off = 128; off > 0; off >>= 1) {
    if (threadIdx.x < off) {
      s1[threadIdx.x] += s1[threadIdx.x + off];
      s2[threadIdx.x] += s2[threadIdx.x + off];
    }
    __syncthreads();
  }
  if (threadIdx.x == 0) {
    float mean = s1[0] * (1.f / (float)NPIX);
    float var  = s2[0] * (1.f / (float)NPIX) - mean * mean;
    stats[o * 2]     = mean;
    stats[o * 2 + 1] = rsqrtf(var + 1e-5f);
  }
}

// ---------------------------------------------------------------------------
// Kernel 4: out = x + gelu_exact(gamma * (y-mean)*rstd + beta)
// ---------------------------------------------------------------------------
__global__ __launch_bounds__(256) void k_final(
    const float* __restrict__ x, const float* __restrict__ y,
    const float* __restrict__ stats,
    const float* __restrict__ gamma, const float* __restrict__ beta,
    float* __restrict__ out)
{
  size_t i = (size_t)blockIdx.x * 256 + threadIdx.x;
  if (i >= (size_t)COUT * NPIX) return;
  int ch = (int)(i >> 10);
  int o  = ch % COUT;
  float mean = stats[o * 2], rstd = stats[o * 2 + 1];
  float yn = (y[i] - mean) * rstd * gamma[o] + beta[o];
  float g  = 0.5f * yn * (1.f + erff(yn * 0.70710678118654752f));
  out[i] = x[i] + g;
}

// ---------------------------------------------------------------------------
extern "C" void kernel_launch(void* const* d_in, const int* in_sizes, int n_in,
                              void* d_out, int out_size, void* d_ws, size_t ws_size,
                              hipStream_t stream) {
  const float* x     = (const float*)d_in[0];
  const float* pw    = (const float*)d_in[1];
  const float* pb    = (const float*)d_in[2];
  const float* ow    = (const float*)d_in[3];
  const float* ob    = (const float*)d_in[4];
  const float* mw    = (const float*)d_in[5];
  const float* mb    = (const float*)d_in[6];
  const float* gamma = (const float*)d_in[7];
  const float* beta  = (const float*)d_in[8];
  float* out = (float*)d_out;

  char* ws = (char*)d_ws;
  const size_t TBL  = (size_t)NPIX * 9 * 4;          // 294912 entries
  const size_t YSZ  = (size_t)COUT * NPIX;           // 6291456
  const size_t WSZ  = (size_t)COUT * KDIM;           // 5308416
  float*  tw    = (float*)ws;
  int*    tix   = (int*)  (ws + TBL * 4);
  float*  yb    = (float*)(ws + 2 * TBL * 4);
  float*  stats = (float*)(ws + 2 * TBL * 4 + YSZ * 4);
  __bf16* whi   = (__bf16*)(ws + 2 * TBL * 4 + YSZ * 4 + 6144);
  __bf16* wlo   = whi + WSZ;

  k_wsplit   <<<(int)((WSZ + 255) / 256), 256, 0, stream>>>(pw, whi, wlo);
  k_offmask  <<<(NPIX * 9 + 255) / 256, 256, 0, stream>>>(x, ow, ob, mw, mb, tw, tix);
  k_dcnv_gemm<<<NPIX / 32, 512, 0, stream>>>(x, whi, wlo, pb, tw, tix, yb);
  k_stats    <<<COUT, 256, 0, stream>>>(yb, stats);
  k_final    <<<(int)((YSZ + 255) / 256), 256, 0, stream>>>(x, yb, stats, gamma, beta, out);
}
